// MaskedMultiHeadAttention_61486751809823
// MI455X (gfx1250) — matrix operand order repaired and checked by reading, not compiled, not run on hardware
//
#include <hip/hip_runtime.h>

#define BB  2
#define SS  2048
#define DDm 1024
#define HHn 16
#define HDD 64

typedef __bf16 bf16_t;
typedef __attribute__((ext_vector_type(16))) __bf16 v16bf;
typedef __attribute__((ext_vector_type(8)))  __bf16 v8bf;
typedef __attribute__((ext_vector_type(8)))  float  v8f;

union Frag16 { v16bf v; v8bf h[2]; };

__device__ __forceinline__ bf16_t to_bf16(float x) {
  unsigned u = __builtin_bit_cast(unsigned, x);
  u += 0x7FFFu + ((u >> 16) & 1u);           // round-to-nearest-even
  unsigned short s = (unsigned short)(u >> 16);
  return __builtin_bit_cast(bf16_t, s);
}

// A-fragment (16x32, 16-bit): lane holds row = lane&15;
// lanes 0-15: K = {0..7, 16..23}; lanes 16-31: K = {8..15, 24..31}.
__device__ __forceinline__ v16bf load_a_frag(const bf16_t* base, int ld, int lane) {
  int r = lane & 15, hi = lane >> 4;
  const bf16_t* p = base + (size_t)r * ld + hi * 8;
  Frag16 f;
  f.h[0] = *(const v8bf*)(p);
  f.h[1] = *(const v8bf*)(p + 16);
  return f.v;
}

// B-fragment (32x16, 16-bit): lane holds column N = lane&15;
// lanes 0-15: K = {0..7, 16..23}; lanes 16-31: K = {8..15, 24..31}.
// Source element (k, n) at base[n*ldn + k]  (K-contiguous columns).
__device__ __forceinline__ v16bf load_b_frag(const bf16_t* base, int ldn, int lane) {
  int n = lane & 15, hi = lane >> 4;
  const bf16_t* p = base + (size_t)n * ldn + hi * 8;
  Frag16 f;
  f.h[0] = *(const v8bf*)(p);
  f.h[1] = *(const v8bf*)(p + 16);
  return f.v;
}

__device__ __forceinline__ v8f wmma_bf16(v16bf a, v16bf b, v8f c) {
  return __builtin_amdgcn_wmma_f32_16x16x32_bf16(false, a, false, b, (short)0, c,
                                                 false, false);
}

__device__ __forceinline__ void wait_lds() {
  asm volatile("" ::: "memory");
#if __has_builtin(__builtin_amdgcn_s_wait_dscnt)
  __builtin_amdgcn_s_wait_dscnt(0);
#else
  asm volatile("s_wait_dscnt 0" ::: "memory");
#endif
  asm volatile("" ::: "memory");
}

// ---------------- convert / transpose kernels ----------------

__global__ void k_cvt_hidden(const float* __restrict__ src, bf16_t* __restrict__ dst,
                             size_t n) {
  for (size_t i = blockIdx.x * (size_t)blockDim.x + threadIdx.x; i < n;
       i += (size_t)gridDim.x * blockDim.x)
    dst[i] = to_bf16(src[i]);
}

// Wq/Wk/Wv [H][D][HD] -> transposed bf16 [H][HD][D]
__global__ void k_cvt_wT(const float* __restrict__ Wq, const float* __restrict__ Wk,
                         const float* __restrict__ Wv, bf16_t* __restrict__ WqT,
                         bf16_t* __restrict__ WkT, bf16_t* __restrict__ WvT) {
  const size_t per = (size_t)HHn * DDm * HDD;
  for (size_t i = blockIdx.x * (size_t)blockDim.x + threadIdx.x; i < 3 * per;
       i += (size_t)gridDim.x * blockDim.x) {
    int sel = (int)(i / per);
    size_t j = i % per;
    int h = (int)(j / ((size_t)DDm * HDD));
    int rem = (int)(j % ((size_t)DDm * HDD));
    int d = rem / HDD, e = rem % HDD;
    const float* W = sel == 0 ? Wq : sel == 1 ? Wk : Wv;
    bf16_t* T = sel == 0 ? WqT : sel == 1 ? WkT : WvT;
    T[((size_t)h * HDD + e) * DDm + d] = to_bf16(W[j]);
  }
}

// Wo [K=D][N=D] -> WoT bf16 [N][K]
__global__ void k_cvt_woT(const float* __restrict__ Wo, bf16_t* __restrict__ WoT) {
  const size_t total = (size_t)DDm * DDm;
  for (size_t i = blockIdx.x * (size_t)blockDim.x + threadIdx.x; i < total;
       i += (size_t)gridDim.x * blockDim.x) {
    int k = (int)(i / DDm), n = (int)(i % DDm);
    WoT[(size_t)n * DDm + k] = to_bf16(Wo[i]);
  }
}

// ---------------- QKV projection (WMMA bf16, 64x64 register block) ---------
// One wave computes a 64-row x 64-col (= whole head) block: 16 accumulators,
// per K-step 8 A-loads + 8 B-loads feed 16 WMMAs (1 load : 1 wmma).
// sel 0/1/2 = q/k/v; v is stored transposed [B,H,HD,S].

__global__ void __launch_bounds__(128)
k_qkv(const bf16_t* __restrict__ hb, const bf16_t* __restrict__ WqT,
      const bf16_t* __restrict__ WkT, const bf16_t* __restrict__ WvT,
      const float* __restrict__ bq, const float* __restrict__ bk,
      const float* __restrict__ bv, bf16_t* __restrict__ qb,
      bf16_t* __restrict__ kbuf, bf16_t* __restrict__ vtb) {
  int lane = threadIdx.x & 31;
  int wid = blockIdx.x * 4 + (threadIdx.x >> 5);
  // wid in [0, 3 * B*H * (S/64)) = [0, 3072)
  int sel = wid >> 10;            // / 1024
  int rem = wid & 1023;
  int mblk = rem & 31;            // S/64 = 32 blocks
  int bh = rem >> 5;              // b*H + h
  int b = bh >> 4, h = bh & 15;
  int m0 = mblk * 64;
  int n = lane & 15, hi = lane >> 4;

  const bf16_t* WT = (sel == 0 ? WqT : sel == 1 ? WkT : WvT) + (size_t)h * HDD * DDm;
  const float* bias_p = (sel == 0 ? bq : sel == 1 ? bk : bv);
  float bias[4];
  for (int ni = 0; ni < 4; ++ni) bias[ni] = bias_p[h * HDD + ni * 16 + n];
  const bf16_t* arow = hb + ((size_t)b * SS + m0) * DDm;

  v8f acc[4][4] = {};
  for (int k0 = 0; k0 < DDm; k0 += 32) {
    if (k0 + 128 < DDm) {
      __builtin_prefetch(arow + k0 + 128, 0, 0);
      __builtin_prefetch(WT + k0 + 128, 0, 0);
    }
    v16bf a[4], w[4];
    for (int mi = 0; mi < 4; ++mi)
      a[mi] = load_a_frag(arow + (size_t)mi * 16 * DDm + k0, DDm, lane);
    for (int ni = 0; ni < 4; ++ni)
      w[ni] = load_b_frag(WT + (size_t)ni * 16 * DDm + k0, DDm, lane);
    for (int mi = 0; mi < 4; ++mi)
      for (int ni = 0; ni < 4; ++ni)
        acc[mi][ni] = wmma_bf16(a[mi], w[ni], acc[mi][ni]);
  }

  if (sel < 2) {
    bf16_t* dst = (sel == 0 ? qb : kbuf) + (size_t)bh * SS * HDD;
    for (int mi = 0; mi < 4; ++mi)
      for (int r = 0; r < 8; ++r) {
        int M = m0 + mi * 16 + r + hi * 8;
        for (int ni = 0; ni < 4; ++ni)
          dst[(size_t)M * HDD + ni * 16 + n] = to_bf16(acc[mi][ni][r] + bias[ni]);
      }
  } else {
    for (int ni = 0; ni < 4; ++ni) {
      bf16_t* col = vtb + ((size_t)bh * HDD + ni * 16 + n) * SS;
      for (int mi = 0; mi < 4; ++mi) {
        v8bf vv;
        for (int r = 0; r < 8; ++r) vv[r] = to_bf16(acc[mi][ni][r] + bias[ni]);
        *(v8bf*)(col + m0 + mi * 16 + hi * 8) = vv;   // contiguous 16B store
      }
    }
  }
}

// ---------------- flash attention core ----------------
// One wave owns a 16-query tile; iterates causal 32-key blocks with online
// softmax; P is transposed C-layout -> A-layout through a per-wave LDS slice.

__global__ void __launch_bounds__(128)
k_attn(const bf16_t* __restrict__ qb, const bf16_t* __restrict__ kbuf,
       const bf16_t* __restrict__ vtb, bf16_t* __restrict__ ctxb) {
  __shared__ bf16_t lds_p[4][16 * 32];
  int lane = threadIdx.x & 31;
  int wslot = threadIdx.x >> 5;
  int wid = blockIdx.x * 4 + wslot;
  int m0 = (wid & 127) * 16;
  int bh = wid >> 7;
  int b = bh >> 4, h = bh & 15;
  int n = lane & 15, hi = lane >> 4;

  const bf16_t* q0  = qb   + (size_t)bh * SS * HDD;
  const bf16_t* kp  = kbuf + (size_t)bh * SS * HDD;
  const bf16_t* vt0 = vtb  + (size_t)bh * HDD * SS;
  bf16_t* ldsw = lds_p[wslot];

  v16bf aq0 = load_a_frag(q0 + (size_t)m0 * HDD + 0, HDD, lane);
  v16bf aq1 = load_a_frag(q0 + (size_t)m0 * HDD + 32, HDD, lane);

  v8f acc0 = {}, acc1 = {}, acc2 = {}, acc3 = {};
  float rmax[8], rsum[8];
  for (int r = 0; r < 8; ++r) { rmax[r] = -1e30f; rsum[r] = 0.f; }

  const float scale = 0.125f;  // 1/sqrt(64)
  int nkb = (m0 + 15) / 32 + 1;
  for (int kbi = 0; kbi < nkb; ++kbi) {
    int t0 = kbi * 32;
    // scores for 32 keys (two 16-col halves), accumulate over 2 feature chunks
    v8f s0 = {}, s1 = {};
    {
      v16bf b00 = load_b_frag(kp + (size_t)t0 * HDD + 0, HDD, lane);
      v16bf b01 = load_b_frag(kp + (size_t)t0 * HDD + 32, HDD, lane);
      s0 = wmma_bf16(aq0, b00, s0);
      s0 = wmma_bf16(aq1, b01, s0);
      v16bf b10 = load_b_frag(kp + (size_t)(t0 + 16) * HDD + 0, HDD, lane);
      v16bf b11 = load_b_frag(kp + (size_t)(t0 + 16) * HDD + 32, HDD, lane);
      s1 = wmma_bf16(aq0, b10, s1);
      s1 = wmma_bf16(aq1, b11, s1);
    }
    // scale + causal mask; per-row tile max
    float p0[8], p1[8], mt[8];
    for (int r = 0; r < 8; ++r) {
      int qrow = m0 + r + hi * 8;
      float x0 = (t0 + n      <= qrow) ? s0[r] * scale : -1e30f;
      float x1 = (t0 + 16 + n <= qrow) ? s1[r] * scale : -1e30f;
      p0[r] = x0; p1[r] = x1;
      mt[r] = fmaxf(x0, x1);
    }
    for (int off = 1; off < 16; off <<= 1)
      for (int r = 0; r < 8; ++r) mt[r] = fmaxf(mt[r], __shfl_xor(mt[r], off, 32));
    float alpha[8];
    for (int r = 0; r < 8; ++r) {
      float mnew = fmaxf(rmax[r], mt[r]);
      alpha[r] = __expf(rmax[r] - mnew);
      rmax[r] = mnew;
      p0[r] = __expf(p0[r] - mnew);
      p1[r] = __expf(p1[r] - mnew);
    }
    float ps[8];
    for (int r = 0; r < 8; ++r) ps[r] = p0[r] + p1[r];
    for (int off = 1; off < 16; off <<= 1)
      for (int r = 0; r < 8; ++r) ps[r] += __shfl_xor(ps[r], off, 32);
    for (int r = 0; r < 8; ++r) {
      rsum[r] = rsum[r] * alpha[r] + ps[r];
      acc0[r] *= alpha[r]; acc1[r] *= alpha[r];
      acc2[r] *= alpha[r]; acc3[r] *= alpha[r];
    }
    // C-layout P -> LDS -> A-layout bf16 fragment (wave-local transpose)
    for (int r = 0; r < 8; ++r) {
      int M = r + hi * 8;
      ldsw[M * 32 + n]      = to_bf16(p0[r]);
      ldsw[M * 32 + 16 + n] = to_bf16(p1[r]);
    }
    wait_lds();
    v16bf ap = load_a_frag(ldsw, 32, lane);
    acc0 = wmma_bf16(ap, load_b_frag(vt0 + (size_t)0  * SS + t0, SS, lane), acc0);
    acc1 = wmma_bf16(ap, load_b_frag(vt0 + (size_t)16 * SS + t0, SS, lane), acc1);
    acc2 = wmma_bf16(ap, load_b_frag(vt0 + (size_t)32 * SS + t0, SS, lane), acc2);
    acc3 = wmma_bf16(ap, load_b_frag(vt0 + (size_t)48 * SS + t0, SS, lane), acc3);
  }

  // normalize and store ctx (bf16, [B,S,D] layout, head h at column h*64)
  bf16_t* crow = ctxb + ((size_t)b * SS + m0) * DDm + h * HDD;
  for (int r = 0; r < 8; ++r) {
    int M = r + hi * 8;
    float inv = 1.f / rsum[r];
    bf16_t* cp = crow + (size_t)M * DDm;
    cp[0 * 16 + n] = to_bf16(acc0[r] * inv);
    cp[1 * 16 + n] = to_bf16(acc1[r] * inv);
    cp[2 * 16 + n] = to_bf16(acc2[r] * inv);
    cp[3 * 16 + n] = to_bf16(acc3[r] * inv);
  }
}

// ---------------- output projection (64x64 register block) ----------------

__global__ void __launch_bounds__(128)
k_proj(const bf16_t* __restrict__ ctxb, const bf16_t* __restrict__ WoT,
       const float* __restrict__ bo, float* __restrict__ out) {
  int lane = threadIdx.x & 31;
  int wid = blockIdx.x * 4 + (threadIdx.x >> 5);
  // wid in [0, (B*S/64) * (D/64)) = [0, 1024)
  int n0 = (wid & 15) * 64;
  int m0 = (wid >> 4) * 64;
  int n = lane & 15, hi = lane >> 4;

  const bf16_t* arow = ctxb + (size_t)m0 * DDm;
  const bf16_t* wcol = WoT + (size_t)n0 * DDm;
  float bias[4];
  for (int ni = 0; ni < 4; ++ni) bias[ni] = bo[n0 + ni * 16 + n];

  v8f acc[4][4] = {};
  for (int k0 = 0; k0 < DDm; k0 += 32) {
    if (k0 + 128 < DDm) {
      __builtin_prefetch(arow + k0 + 128, 0, 0);
      __builtin_prefetch(wcol + k0 + 128, 0, 0);
    }
    v16bf a[4], w[4];
    for (int mi = 0; mi < 4; ++mi)
      a[mi] = load_a_frag(arow + (size_t)mi * 16 * DDm + k0, DDm, lane);
    for (int ni = 0; ni < 4; ++ni)
      w[ni] = load_b_frag(wcol + (size_t)ni * 16 * DDm + k0, DDm, lane);
    for (int mi = 0; mi < 4; ++mi)
      for (int ni = 0; ni < 4; ++ni)
        acc[mi][ni] = wmma_bf16(a[mi], w[ni], acc[mi][ni]);
  }

  for (int mi = 0; mi < 4; ++mi)
    for (int r = 0; r < 8; ++r) {
      int M = m0 + mi * 16 + r + hi * 8;
      for (int ni = 0; ni < 4; ++ni)
        out[(size_t)M * DDm + n0 + ni * 16 + n] = acc[mi][ni][r] + bias[ni];
    }
}

// ---------------- host launcher ----------------

extern "C" void kernel_launch(void* const* d_in, const int* in_sizes, int n_in,
                              void* d_out, int out_size, void* d_ws, size_t ws_size,
                              hipStream_t stream) {
  (void)in_sizes; (void)n_in; (void)out_size; (void)ws_size;
  const float* hidden = (const float*)d_in[0];
  const float* Wq = (const float*)d_in[1];
  const float* bq = (const float*)d_in[2];
  const float* Wk = (const float*)d_in[3];
  const float* bk = (const float*)d_in[4];
  const float* Wv = (const float*)d_in[5];
  const float* bv = (const float*)d_in[6];
  const float* Wo = (const float*)d_in[7];
  const float* bo = (const float*)d_in[8];
  float* out = (float*)d_out;

  char* ws = (char*)d_ws;
  const size_t MB = (size_t)1 << 20;
  bf16_t* hb   = (bf16_t*)(ws + 0);        //  8 MB bf16 hidden
  bf16_t* WqT  = (bf16_t*)(ws + 8 * MB);   //  2 MB
  bf16_t* WkT  = (bf16_t*)(ws + 10 * MB);  //  2 MB
  bf16_t* WvT  = (bf16_t*)(ws + 12 * MB);  //  2 MB
  bf16_t* WoT  = (bf16_t*)(ws + 14 * MB);  //  2 MB
  bf16_t* qb   = (bf16_t*)(ws + 16 * MB);  //  8 MB  [B,H,S,HD]
  bf16_t* kb   = (bf16_t*)(ws + 24 * MB);  //  8 MB  [B,H,S,HD]
  bf16_t* vtb  = (bf16_t*)(ws + 32 * MB);  //  8 MB  [B,H,HD,S]
  bf16_t* ctxb = (bf16_t*)(ws + 40 * MB);  //  8 MB  [B,S,D]

  k_cvt_hidden<<<2048, 256, 0, stream>>>(hidden, hb, (size_t)BB * SS * DDm);
  k_cvt_wT<<<1024, 256, 0, stream>>>(Wq, Wk, Wv, WqT, WkT, WvT);
  k_cvt_woT<<<512, 256, 0, stream>>>(Wo, WoT);

  // 3 * B*H * (S/64) = 3072 waves, 4 waves/block
  k_qkv<<<768, 128, 0, stream>>>(hb, WqT, WkT, WvT, bq, bk, bv, qb, kb, vtb);
  // B*H*(S/16) = 4096 waves
  k_attn<<<1024, 128, 0, stream>>>(qb, kb, vtb, ctxb);
  // (B*S/64)*(D/64) = 1024 waves
  k_proj<<<256, 128, 0, stream>>>(ctxb, WoT, bo, out);
}
